// LQE_38938173505554
// MI455X (gfx1250) — matrix operand order repaired and checked by reading, not compiled, not run on hardware
//
#include <hip/hip_runtime.h>

#define REG_BINS 33
#define NCLS     80
#define FEAT     20            // 4*(K+1)
#define HID      64
#define BPB      64            // boxes per block
#define THREADS  128           // 4 waves (wave32)
#define CORN     (4 * REG_BINS)   // 132 floats per box

typedef __attribute__((ext_vector_type(16))) _Float16 v16h;
typedef __attribute__((ext_vector_type(8)))  float    v8f;

__global__ __launch_bounds__(THREADS)
void lqe_kernel(const float* __restrict__ scores,   // [NB, 80]
                const float* __restrict__ corners,  // [NB, 132]
                const float* __restrict__ W1,       // [64, 20] row-major
                const float* __restrict__ b1,       // [64]
                const float* __restrict__ W2,       // [64]  (1x64)
                const float* __restrict__ b2,       // [1]
                float* __restrict__ out)            // [NB, 80]
{
    __shared__ float sCorn[BPB * CORN];   // 33792 B
    __shared__ float sStat[BPB * FEAT];   //  5120 B
    __shared__ float sQual[BPB];          //   256 B

    const int tid  = threadIdx.x;
    const int lane = tid & 31;
    const int wave = tid >> 5;
    const int nLo  = lane & 15;
    const int g    = lane >> 4;           // half-wave group
    const int chunkBase = blockIdx.x * BPB;   // first box of this block

    // ---------- per-wave constant setup: W1^T as f16 B-tiles, per-lane w2/b1 ----------
    // B k order used here (contiguous, same as A): lanes 0-15 hold K=0..15 (elem e -> K=e),
    //                           lanes 16-31 hold K=16..31 (elem e -> K=16+e), col N = lane%16.
    v16h Btile[4];
    float w2l[4], b1l[4];
#pragma unroll
    for (int t = 0; t < 4; ++t) {
        const int n = nLo + 16 * t;                  // hidden unit index
#pragma unroll
        for (int e = 0; e < 16; ++e) {
            const int k = e + 16 * g;                // feature index (pad K=20..31 with 0)
            const float w = (k < FEAT) ? W1[n * FEAT + k] : 0.0f;
            Btile[t][e] = (_Float16)w;
        }
        w2l[t] = W2[n];
        b1l[t] = b1[n];
    }
    const float b2v = b2[0];

    // ---------- stage pred_corners -> LDS, fully coalesced b128 ----------
    {
        const float4* gc4 = (const float4*)(corners + (size_t)chunkBase * CORN);
        float4* sc4 = (float4*)sCorn;
        const int NC4 = BPB * CORN / 4;              // 2112
        for (int i = tid; i < NC4; i += THREADS)
            sc4[i] = gc4[i];
    }
    __syncthreads();

    // ---------- phase 1: softmax + top4 per (box, side); 256 sides, 2/thread ----------
#pragma unroll
    for (int rep = 0; rep < 2; ++rep) {
        const int sid = tid + rep * THREADS;         // 0..255
        const int bl  = sid >> 2;                    // local box
        const int s   = sid & 3;                     // side
        const float* v = &sCorn[bl * CORN + s * REG_BINS];

        float mx = v[0];
#pragma unroll
        for (int j = 1; j < REG_BINS; ++j) mx = fmaxf(mx, v[j]);

        float sum = 0.0f;
        float t0 = -1e30f, t1 = -1e30f, t2 = -1e30f, t3 = -1e30f;
#pragma unroll
        for (int j = 0; j < REG_BINS; ++j) {
            const float x = v[j];
            sum += __expf(x - mx);
            // sorted-descending insert of x into {t0..t3}; g0 => g1 => g2 => g3
            const bool g0 = x > t0, g1 = x > t1, g2 = x > t2, g3 = x > t3;
            const float n3 = g3 ? (g2 ? t2 : x) : t3;
            const float n2 = g2 ? (g1 ? t1 : x) : t2;
            const float n1 = g1 ? (g0 ? t0 : x) : t1;
            const float n0 = g0 ? x : t0;
            t0 = n0; t1 = n1; t2 = n2; t3 = n3;
        }
        const float inv = __builtin_amdgcn_rcpf(sum);
        const float p0 = __expf(t0 - mx) * inv;
        const float p1 = __expf(t1 - mx) * inv;
        const float p2 = __expf(t2 - mx) * inv;
        const float p3 = __expf(t3 - mx) * inv;
        float* st = &sStat[bl * FEAT + s * 5];
        st[0] = p0; st[1] = p1; st[2] = p2; st[3] = p3;
        st[4] = 0.25f * (p0 + p1 + p2 + p3);
    }
    __syncthreads();

    // ---------- phase 2: MLP layer 1 via WMMA; wave w owns boxes [16w, 16w+16) ----------
    {
        const int boxL = wave * 16 + nLo;            // local box supplying this lane's A rows
        // A k order used here (contiguous, same as B): lanes 0-15: elem e -> K = e;
        //                           lanes 16-31: elem e -> K = 16+e. Row M = lane%16.
        v16h A;
#pragma unroll
        for (int e = 0; e < 16; ++e) {
            const int k = e + 16 * g;
            const float x = (k < FEAT) ? sStat[boxL * FEAT + k] : 0.0f;
            A[e] = (_Float16)x;
        }

        v8f acc[4];
#pragma unroll
        for (int t = 0; t < 4; ++t) {
            v8f c;
#pragma unroll
            for (int r = 0; r < 8; ++r) c[r] = b1l[t];   // bias depends only on N
            acc[t] = __builtin_amdgcn_wmma_f32_16x16x32_f16(
                false, A, false, Btile[t], (short)0, c, false, false);
        }

        // ReLU + layer-2 per-lane partial dot: lane holds cols N = nLo+16t for rows r(+8g)
        float p[8];
#pragma unroll
        for (int r = 0; r < 8; ++r) {
            float s = 0.0f;
#pragma unroll
            for (int t = 0; t < 4; ++t)
                s += fmaxf(acc[t][r], 0.0f) * w2l[t];
            p[r] = s;
        }
        // butterfly-reduce across the 16 lanes of each half-wave group
#pragma unroll
        for (int mask = 1; mask <= 8; mask <<= 1) {
#pragma unroll
            for (int r = 0; r < 8; ++r)
                p[r] += __shfl_xor(p[r], mask, 32);
        }
        if (nLo == 0) {
#pragma unroll
            for (int r = 0; r < 8; ++r)
                sQual[wave * 16 + 8 * g + r] = p[r] + b2v;
        }
    }
    __syncthreads();

    // ---------- phase 3: out = scores + quality, coalesced float4 stream ----------
    {
        const float4* gs4 = (const float4*)(scores + (size_t)chunkBase * NCLS);
        float4* go4 = (float4*)out + (size_t)chunkBase * (NCLS / 4);
        const int NO4 = BPB * NCLS / 4;              // 1280
        for (int i = tid; i < NO4; i += THREADS) {
            const int bl = i / (NCLS / 4);
            const float q = sQual[bl];
            float4 sv = gs4[i];
            sv.x += q; sv.y += q; sv.z += q; sv.w += q;
            go4[i] = sv;
        }
    }
}

extern "C" void kernel_launch(void* const* d_in, const int* in_sizes, int n_in,
                              void* d_out, int out_size, void* d_ws, size_t ws_size,
                              hipStream_t stream) {
    const float* scores  = (const float*)d_in[0];  // [32, 8400, 80]
    const float* corners = (const float*)d_in[1];  // [32, 8400, 132]
    const float* W1      = (const float*)d_in[2];  // [64, 20]
    const float* b1      = (const float*)d_in[3];  // [64]
    const float* W2      = (const float*)d_in[4];  // [1, 64]
    const float* b2      = (const float*)d_in[5];  // [1]
    float* out = (float*)d_out;

    const int NB = in_sizes[0] / NCLS;             // 268800 boxes
    const int nBlocks = NB / BPB;                  // 4200
    lqe_kernel<<<nBlocks, THREADS, 0, stream>>>(scores, corners, W1, b1, W2, b2, out);
}
